// SAM3Segmenter_88167088652461
// MI455X (gfx1250) — 8-bit matrix operand order repaired and checked by reading, compiled, not run on hardware
//
#include <hip/hip_runtime.h>

typedef int v8i __attribute__((ext_vector_type(8)));

#define NQ    96
#define HWSZ  (512*512)
#define KC    512
#define KCD   (KC/4)
#define NCHUNK (HWSZ/KC)
#define NTILE 6
#define TILES (NTILE*NTILE)
#define NPROM 8

__global__ void sam3_zero_ws(int* __restrict__ ws) {
    int i = blockIdx.x * blockDim.x + threadIdx.x;
    if (i < NQ * NQ) ws[i] = 0;
}

__global__ __launch_bounds__(256)
void sam3_mask_gram(const float* __restrict__ logits, int* __restrict__ inter) {
    __shared__ unsigned smem[NQ * KCD];

    const int t  = threadIdx.x;
    const int l  = t & 31;
    const int w  = t >> 5;
    const int lo = l & 15;
    const int hi = (l >> 4);

    int tii[5], tjj[5];
    v8i acc[5] = {};
    #pragma unroll
    for (int s = 0; s < 5; ++s) {
        int tt = w + 8 * s;
        int vt = (tt < TILES) ? tt : 0;
        tii[s] = (vt / NTILE) * 16;
        tjj[s] = (vt % NTILE) * 16;
    }

    for (int c = blockIdx.x; c < NCHUNK; c += gridDim.x) {
        __syncthreads();
        const size_t kbase = (size_t)c * KC;
        for (int e = t; e < NQ * KCD; e += 256) {
            const int row = e / KCD;
            const int dw  = e % KCD;
            const float4 f =
                ((const float4*)logits)[(((size_t)row * HWSZ + kbase) >> 2) + dw];
            unsigned b = (f.x > 0.f ? 1u : 0u)
                       | (f.y > 0.f ? 1u : 0u) << 8
                       | (f.z > 0.f ? 1u : 0u) << 16
                       | (f.w > 0.f ? 1u : 0u) << 24;
            smem[e] = b;
        }
        __syncthreads();

        for (int ks = 0; ks < KCD; ks += 16) {
            #pragma unroll
            for (int s = 0; s < 5; ++s) {
                const int rowA = tii[s] + lo;
                const int rowB = tjj[s] + lo;
                v8i a, b;
                #pragma unroll
                for (int v = 0; v < 8; ++v) {
                    const int dA = (hi ? 2 : 0) + (v & 1) + ((v >> 1) & 1) * 4
                                 + (v >> 2) * 8;
                    const int dB = (hi ? 2 : 0) + (v & 1) + ((v >> 1) & 1) * 4
                                 + (v >> 2) * 8;
                    a[v] = (int)smem[rowA * KCD + ks + dA];
                    b[v] = (int)smem[rowB * KCD + ks + dB];
                }
                acc[s] = __builtin_amdgcn_wmma_i32_16x16x64_iu8(
                     false, a,  false, b,
                    acc[s],  false,  false);
            }
        }
    }

    #pragma unroll
    for (int s = 0; s < 5; ++s) {
        if (w + 8 * s < TILES) {
            #pragma unroll
            for (int r = 0; r < 8; ++r) {
                const int gi = tii[s] + r + hi * 8;
                const int gj = tjj[s] + lo;
                atomicAdd(&inter[gi * NQ + gj], acc[s][r]);
            }
        }
    }
}

__global__ __launch_bounds__(128)
void sam3_finalize(const int* __restrict__ inter,
                   const float* __restrict__ scores,
                   const float* __restrict__ pboxes,
                   const float* __restrict__ qboxes,
                   float* __restrict__ out) {
    __shared__ float s_iou[NQ * NQ];
    __shared__ float s_sc[NQ];
    __shared__ int   s_order[NQ];
    __shared__ int   s_keep[NQ];
    __shared__ float s_comb[NPROM * NQ];
    __shared__ int   s_ord[NPROM * NQ];
    __shared__ int   s_assign[NPROM];
    __shared__ int   s_used[NQ];
    __shared__ int   s_any;

    const int t = threadIdx.x;
    if (t < NQ) s_sc[t] = scores[t];
    if (t == 0) s_any = 0;
    __syncthreads();

    for (int e = t; e < NQ * NQ; e += 128) {
        const int i = e / NQ, j = e % NQ;
        const float in_ = (float)inter[e];
        const float un  = (float)inter[i * NQ + i] + (float)inter[j * NQ + j] - in_;
        s_iou[e] = (un > 0.f) ? in_ / un : 0.f;
    }

    if (t < NQ) {
        const float si = s_sc[t];
        int r = 0;
        for (int j = 0; j < NQ; ++j) {
            const float sj = s_sc[j];
            if (sj > si || (sj == si && j < t)) ++r;
        }
        s_order[r] = t;
        if (si >= 0.5f) atomicOr(&s_any, 1);
    }

    for (int e = t; e < NPROM * NQ; e += 128) {
        const int p = e / NQ, q = e % NQ;
        const float ax0 = pboxes[p*4+0], ay0 = pboxes[p*4+1];
        const float ax1 = pboxes[p*4+2], ay1 = pboxes[p*4+3];
        const float bx0 = qboxes[q*4+0], by0 = qboxes[q*4+1];
        const float bx1 = qboxes[q*4+2], by1 = qboxes[q*4+3];
        const float wv  = fmaxf(fminf(ax1, bx1) - fmaxf(ax0, bx0), 0.f);
        const float hv  = fmaxf(fminf(ay1, by1) - fmaxf(ay0, by0), 0.f);
        const float in_ = wv * hv;
        const float aa  = fmaxf(ax1 - ax0, 0.f) * fmaxf(ay1 - ay0, 0.f);
        const float ab  = fmaxf(bx1 - bx0, 0.f) * fmaxf(by1 - by0, 0.f);
        const float un  = aa + ab - in_;
        s_comb[e] = ((un > 0.f) ? in_ / un : 0.f) + 0.05f * s_sc[q];
    }
    __syncthreads();

    for (int e = t; e < NPROM * NQ; e += 128) {
        const float ce = s_comb[e];
        int r = 0;
        for (int j = 0; j < NPROM * NQ; ++j) {
            const float cj = s_comb[j];
            if (cj > ce || (cj == ce && j < e)) ++r;
        }
        s_ord[r] = e;
    }
    if (t < NQ)    s_used[t]   = 0;
    if (t < NPROM) s_assign[t] = -1;
    __syncthreads();

    if (t == 0) {
        const int any = s_any;
        for (int i = 0; i < NQ; ++i) {
            const int oi = s_order[i];
            int sup = 0;
            for (int j = 0; j < i; ++j) {
                const int oj = s_order[j];
                if (s_keep[oj] && s_iou[oi * NQ + oj] > 0.5f) { sup = 1; break; }
            }
            const int elig = any ? (s_sc[oi] >= 0.5f ? 1 : 0) : 1;
            s_keep[oi] = elig && !sup;
        }
        int done = 0;
        for (int f = 0; f < NPROM * NQ && done < NPROM; ++f) {
            const int e = s_ord[f];
            const int p = e / NQ, q = e % NQ;
            if (s_assign[p] < 0 && !s_used[q]) {
                s_assign[p] = q; s_used[q] = 1; ++done;
            }
        }
        for (int p = 0; p < NPROM; ++p) {
            if (s_assign[p] < 0) {
                float best = -3.402823466e38f; int bq = 0;
                for (int j = 0; j < NQ; ++j) {
                    const float v = s_used[j] ? -3.402823466e38f : s_sc[j];
                    if (v > best) { best = v; bq = j; }
                }
                s_assign[p] = bq; s_used[bq] = 1;
            }
        }
    }
    __syncthreads();

    if (t < NQ)    out[t]      = s_keep[t] ? s_sc[t] : 0.f;
    if (t < NPROM) out[NQ + t] = s_comb[t * NQ + s_assign[t]];
}

extern "C" void kernel_launch(void* const* d_in, const int* in_sizes, int n_in,
                              void* d_out, int out_size, void* d_ws, size_t ws_size,
                              hipStream_t stream) {
    const float* mask_logits = (const float*)d_in[0];
    const float* scores      = (const float*)d_in[1];
    const float* pboxes      = (const float*)d_in[2];
    const float* qboxes      = (const float*)d_in[3];
    float*       out         = (float*)d_out;
    int*         inter       = (int*)d_ws;

    sam3_zero_ws<<<(NQ * NQ + 255) / 256, 256, 0, stream>>>(inter);
    sam3_mask_gram<<<128, 256, 0, stream>>>(mask_logits, inter);
    sam3_finalize<<<1, 128, 0, stream>>>(inter, scores, pboxes, qboxes, out);
}
